// Decoder_51582557225696
// MI455X (gfx1250) — matrix operand order repaired and checked by reading, not compiled, not run on hardware
//
#include <hip/hip_runtime.h>

// ---------------- types ----------------
typedef __bf16 bf16;
typedef __bf16 v16bf __attribute__((ext_vector_type(16)));
typedef __bf16 v8bf  __attribute__((ext_vector_type(8)));
typedef float  v8f   __attribute__((ext_vector_type(8)));
typedef float  v2f   __attribute__((ext_vector_type(2)));
typedef int    v4i   __attribute__((vector_size(16)));   // matches builtin param type

#define WMMA_BF16(A,B,C) __builtin_amdgcn_wmma_f32_16x16x32_bf16(false,(A),false,(B),(short)0,(C),false,false)
#define WMMA_F32(A,B,C)  __builtin_amdgcn_wmma_f32_16x16x4_f32(false,(A),false,(B),(short)0,(C),false,false)

// ---------------- CDNA5 async global->LDS staging (guarded) ----------------
#if defined(__HIP_DEVICE_COMPILE__) && __has_builtin(__builtin_amdgcn_global_load_async_to_lds_b128) && __has_builtin(__builtin_amdgcn_s_wait_asynccnt)
#define USE_ASYNC_LDS 1
#else
#define USE_ASYNC_LDS 0
#endif

#if USE_ASYNC_LDS
__device__ __forceinline__ void async_cp16(const bf16* g, bf16* l) {
  __builtin_amdgcn_global_load_async_to_lds_b128(
      (__attribute__((address_space(1))) v4i*)(uintptr_t)g,
      (__attribute__((address_space(3))) v4i*)(uint32_t)(uintptr_t)l,
      0, 0);
}
#define ASYNC_DRAIN() __builtin_amdgcn_s_wait_asynccnt(0)
#else
__device__ __forceinline__ void async_cp16(const bf16* g, bf16* l) {
  *(v8bf*)l = *(const v8bf*)g;   // fallback: load + ds_store
}
#define ASYNC_DRAIN() ((void)0)
#endif

// ---------------- problem constants (fixed by reference) ----------------
#define S_LEN   2048
#define D_MODEL 2048
#define NH      16
#define NKVH    8
#define HD      128
#define FF_DIM  5632
#define PREFIX  2048
#define KV_LEN  (PREFIX + S_LEN)       // 4096
#define QDIM    (NH*HD)                // 2048
#define KVDIM   (NKVH*HD)              // 1024
#define LOG1E4  9.210340371976184f     // ln(10000)

// ======================================================================
// Streaming fp32 -> (bf16 hi, bf16 lo) split.  n must be a multiple of 8.
// ======================================================================
__global__ void __launch_bounds__(256) split_f32_kernel(
    const float* __restrict__ W, bf16* __restrict__ Whi, bf16* __restrict__ Wlo,
    long long n) {
  long long i = ((long long)blockIdx.x * blockDim.x + threadIdx.x) * 8;
  if (i >= n) return;
  const float4* p = (const float4*)(W + i);
  float4 f0 = p[0], f1 = p[1];
  float vv[8] = {f0.x, f0.y, f0.z, f0.w, f1.x, f1.y, f1.z, f1.w};
  v8bf hi, lo;
  #pragma unroll
  for (int j = 0; j < 8; j++) {
    bf16 h = (bf16)vv[j];
    hi[j] = h;
    lo[j] = (bf16)(vv[j] - (float)h);
  }
  *(v8bf*)(Whi + i) = hi;
  *(v8bf*)(Wlo + i) = lo;
}

// ======================================================================
// RMSNorm + bf16 hi/lo split
// ======================================================================
__global__ void __launch_bounds__(256) rmsnorm_split_kernel(
    const float* __restrict__ x, const float* __restrict__ w,
    bf16* __restrict__ out_hi, bf16* __restrict__ out_lo, int K) {
  int row = blockIdx.x;
  const float* xr = x + (size_t)row * K;
  float ss = 0.f;
  for (int k = threadIdx.x; k < K; k += blockDim.x) { float v = xr[k]; ss += v * v; }
  #pragma unroll
  for (int off = 16; off; off >>= 1) ss += __shfl_xor(ss, off, 32);
  __shared__ float wsum[8];
  int wid = threadIdx.x >> 5, ln = threadIdx.x & 31;
  if (ln == 0) wsum[wid] = ss;
  __syncthreads();
  if (wid == 0) {
    float v = (ln < 8) ? wsum[ln] : 0.f;
    #pragma unroll
    for (int off = 4; off; off >>= 1) v += __shfl_xor(v, off, 32);
    if (ln == 0) wsum[0] = v;
  }
  __syncthreads();
  float rstd = rsqrtf(wsum[0] / (float)K + 1e-5f);
  for (int k = threadIdx.x; k < K; k += blockDim.x) {
    float v = xr[k] * rstd * w[k];
    bf16 hi = (bf16)v;
    bf16 lo = (bf16)(v - (float)hi);
    out_hi[(size_t)row * K + k] = hi;
    out_lo[(size_t)row * K + k] = lo;
  }
}

// ======================================================================
// Fragment helpers (V_WMMA_F32_16X16X32_BF16 layouts)
// ======================================================================
__device__ __forceinline__ v16bf load_a_frag(const bf16* rowk0, int lhalf) {
  v8bf a0 = *(const v8bf*)(rowk0 + 8 * lhalf);
  v8bf a1 = *(const v8bf*)(rowk0 + 8 * lhalf + 16);
  v16bf r;
  #pragma unroll
  for (int i = 0; i < 8; i++) { r[i] = a0[i]; r[i + 8] = a1[i]; }
  return r;
}

__device__ __forceinline__ v16bf load_b_frag(const bf16* wk16) {
  v8bf b0 = *(const v8bf*)(wk16);
  v8bf b1 = *(const v8bf*)(wk16 + 16);
  v16bf r;
  #pragma unroll
  for (int i = 0; i < 8; i++) { r[i] = b0[i]; r[i + 8] = b1[i]; }
  return r;
}

// ======================================================================
// Generic GEMM:  Y[M,N] = Xsplit[M,K] @ Wsplit[N,K]^T (+ optional residual)
// block = 256 thr (8 waves, 2x4), block tile 64x128, wave tile 32x32.
// A tile staged global->LDS with async copies (double buffered);
// split-bf16: acc += Ahi*Bhi + Ahi*Blo + Alo*Bhi  (~fp32 accuracy)
// ======================================================================
__global__ void __launch_bounds__(256) gemm_bf16s_kernel(
    const bf16* __restrict__ Xhi, const bf16* __restrict__ Xlo,
    const bf16* __restrict__ Whi, const bf16* __restrict__ Wlo,
    const float* __restrict__ Res, float* __restrict__ Y,
    int M, int N, int K) {
  __shared__ bf16 AsHi[2][64 * 32];
  __shared__ bf16 AsLo[2][64 * 32];

  int tid = threadIdx.x;
  int wid = tid >> 5, l = tid & 31, lm = l & 15, lhalf = l >> 4;
  int wm = wid >> 2, wn = wid & 3;
  int bm = blockIdx.y * 64 + wm * 32;
  int bn = blockIdx.x * 128 + wn * 32;
  int bmBase = blockIdx.y * 64;

  // staging: thread -> (row 0..63, 8-elem chunk 0..3) of the 64x32 A tile
  int srow = tid >> 2;
  int sch  = (tid & 3) * 8;
  size_t gArow = (size_t)(bmBase + srow) * K + sch;
  int sidx = srow * 32 + sch;

  size_t wrow0 = (size_t)(bn + lm) * K + 8 * lhalf;
  size_t wrow1 = (size_t)(bn + 16 + lm) * K + 8 * lhalf;

  // LDS fragment bases for this wave's two M sub-tiles
  int ar0 = ((wm * 32) + lm) * 32;
  int ar1 = ((wm * 32) + 16 + lm) * 32;

  // prologue: stage k0 = 0 into buffer 0
  async_cp16(Xhi + gArow, &AsHi[0][sidx]);
  async_cp16(Xlo + gArow, &AsLo[0][sidx]);

  v8f acc[2][2] = {};
  int buf = 0;
  for (int k0 = 0; k0 < K; k0 += 32) {
    ASYNC_DRAIN();
    __syncthreads();
    if (k0 + 32 < K) {   // stage next K chunk into the other buffer
      async_cp16(Xhi + gArow + k0 + 32, &AsHi[buf ^ 1][sidx]);
      async_cp16(Xlo + gArow + k0 + 32, &AsLo[buf ^ 1][sidx]);
    }
    v16bf ahi[2], alo[2], bhi[2], blo[2];
    ahi[0] = load_a_frag(&AsHi[buf][ar0], lhalf);
    alo[0] = load_a_frag(&AsLo[buf][ar0], lhalf);
    ahi[1] = load_a_frag(&AsHi[buf][ar1], lhalf);
    alo[1] = load_a_frag(&AsLo[buf][ar1], lhalf);
    bhi[0] = load_b_frag(Whi + wrow0 + k0);
    blo[0] = load_b_frag(Wlo + wrow0 + k0);
    bhi[1] = load_b_frag(Whi + wrow1 + k0);
    blo[1] = load_b_frag(Wlo + wrow1 + k0);
    __builtin_prefetch((const void*)(Whi + wrow0 + k0 + 32), 0, 3);
    __builtin_prefetch((const void*)(Whi + wrow1 + k0 + 32), 0, 3);
    #pragma unroll
    for (int i = 0; i < 2; i++)
      #pragma unroll
      for (int j = 0; j < 2; j++) {
        acc[i][j] = WMMA_BF16(ahi[i], bhi[j], acc[i][j]);
        acc[i][j] = WMMA_BF16(ahi[i], blo[j], acc[i][j]);
        acc[i][j] = WMMA_BF16(alo[i], bhi[j], acc[i][j]);
      }
    buf ^= 1;
  }
  if (Res != nullptr) {
    #pragma unroll
    for (int i = 0; i < 2; i++)
      #pragma unroll
      for (int j = 0; j < 2; j++)
        #pragma unroll
        for (int r = 0; r < 8; r++) {
          int m = bm + i * 16 + r + 8 * lhalf;
          int n = bn + j * 16 + lm;
          size_t idx = (size_t)m * N + n;
          Y[idx] = acc[i][j][r] + Res[idx];
        }
  } else {
    #pragma unroll
    for (int i = 0; i < 2; i++)
      #pragma unroll
      for (int j = 0; j < 2; j++)
        #pragma unroll
        for (int r = 0; r < 8; r++) {
          int m = bm + i * 16 + r + 8 * lhalf;
          int n = bn + j * 16 + lm;
          Y[(size_t)m * N + n] = acc[i][j][r];
        }
  }
}

// ======================================================================
// Fused gate+up GEMM with SiLU(g)*u epilogue, output split to bf16 hi/lo
// ======================================================================
__global__ void __launch_bounds__(256) gemm_gateup_kernel(
    const bf16* __restrict__ Xhi, const bf16* __restrict__ Xlo,
    const bf16* __restrict__ Wghi, const bf16* __restrict__ Wglo,
    const bf16* __restrict__ Wuhi, const bf16* __restrict__ Wulo,
    bf16* __restrict__ Fhi, bf16* __restrict__ Flo, int M, int N, int K) {
  __shared__ bf16 AsHi[2][64 * 32];
  __shared__ bf16 AsLo[2][64 * 32];

  int tid = threadIdx.x;
  int wid = tid >> 5, l = tid & 31, lm = l & 15, lhalf = l >> 4;
  int wm = wid >> 2, wn = wid & 3;
  int bm = blockIdx.y * 64 + wm * 32;
  int bn = blockIdx.x * 128 + wn * 32;
  int bmBase = blockIdx.y * 64;

  int srow = tid >> 2;
  int sch  = (tid & 3) * 8;
  size_t gArow = (size_t)(bmBase + srow) * K + sch;
  int sidx = srow * 32 + sch;

  size_t wrow0 = (size_t)(bn + lm) * K + 8 * lhalf;
  size_t wrow1 = (size_t)(bn + 16 + lm) * K + 8 * lhalf;
  int ar0 = ((wm * 32) + lm) * 32;
  int ar1 = ((wm * 32) + 16 + lm) * 32;

  async_cp16(Xhi + gArow, &AsHi[0][sidx]);
  async_cp16(Xlo + gArow, &AsLo[0][sidx]);

  v8f accg[2][2] = {}, accu[2][2] = {};
  int buf = 0;
  for (int k0 = 0; k0 < K; k0 += 32) {
    ASYNC_DRAIN();
    __syncthreads();
    if (k0 + 32 < K) {
      async_cp16(Xhi + gArow + k0 + 32, &AsHi[buf ^ 1][sidx]);
      async_cp16(Xlo + gArow + k0 + 32, &AsLo[buf ^ 1][sidx]);
    }
    v16bf ahi[2], alo[2], bghi[2], bglo[2], buhi[2], bulo[2];
    ahi[0] = load_a_frag(&AsHi[buf][ar0], lhalf);
    alo[0] = load_a_frag(&AsLo[buf][ar0], lhalf);
    ahi[1] = load_a_frag(&AsHi[buf][ar1], lhalf);
    alo[1] = load_a_frag(&AsLo[buf][ar1], lhalf);
    bghi[0] = load_b_frag(Wghi + wrow0 + k0);
    bglo[0] = load_b_frag(Wglo + wrow0 + k0);
    bghi[1] = load_b_frag(Wghi + wrow1 + k0);
    bglo[1] = load_b_frag(Wglo + wrow1 + k0);
    buhi[0] = load_b_frag(Wuhi + wrow0 + k0);
    bulo[0] = load_b_frag(Wulo + wrow0 + k0);
    buhi[1] = load_b_frag(Wuhi + wrow1 + k0);
    bulo[1] = load_b_frag(Wulo + wrow1 + k0);
    #pragma unroll
    for (int i = 0; i < 2; i++)
      #pragma unroll
      for (int j = 0; j < 2; j++) {
        accg[i][j] = WMMA_BF16(ahi[i], bghi[j], accg[i][j]);
        accg[i][j] = WMMA_BF16(ahi[i], bglo[j], accg[i][j]);
        accg[i][j] = WMMA_BF16(alo[i], bghi[j], accg[i][j]);
        accu[i][j] = WMMA_BF16(ahi[i], buhi[j], accu[i][j]);
        accu[i][j] = WMMA_BF16(ahi[i], bulo[j], accu[i][j]);
        accu[i][j] = WMMA_BF16(alo[i], buhi[j], accu[i][j]);
      }
    buf ^= 1;
  }
  #pragma unroll
  for (int i = 0; i < 2; i++)
    #pragma unroll
    for (int j = 0; j < 2; j++)
      #pragma unroll
      for (int r = 0; r < 8; r++) {
        int m = bm + i * 16 + r + 8 * lhalf;
        int n = bn + j * 16 + lm;
        float g = accg[i][j][r], u = accu[i][j][r];
        float v = (g / (1.f + __expf(-g))) * u;   // SiLU(g) * u
        size_t idx = (size_t)m * N + n;
        bf16 hi = (bf16)v;
        Fhi[idx] = hi;
        Flo[idx] = (bf16)(v - (float)hi);
      }
}

// ======================================================================
// RoPE on Q in place:  q layout [S, NH*HD], position = prefix + s
// ======================================================================
__global__ void __launch_bounds__(256) rope_q_kernel(float* __restrict__ q) {
  int s = blockIdx.x;
  float pos = (float)(PREFIX + s);
  for (int p = threadIdx.x; p < NH * (HD / 2); p += blockDim.x) {
    int h = p / (HD / 2), i = p % (HD / 2);
    float invf = expf(-(2.f * (float)i / (float)HD) * LOG1E4);
    float sn, cs;
    sincosf(pos * invf, &sn, &cs);
    size_t i0 = (size_t)s * QDIM + h * HD + i;
    size_t i1 = i0 + HD / 2;
    float x0 = q[i0], x1 = q[i1];
    q[i0] = x0 * cs - x1 * sn;
    q[i1] = x1 * cs + x0 * sn;
  }
}

// ======================================================================
// Build roped keys [NH, KV_LEN, HD] and concatenated vals (GQA expand)
// ======================================================================
__global__ void __launch_bounds__(64) build_kv_kernel(
    const float* __restrict__ kc, const float* __restrict__ vc,
    const float* __restrict__ knew, const float* __restrict__ vnew,
    float* __restrict__ keys, float* __restrict__ vals) {
  int p = blockIdx.x;            // 0..KV_LEN-1
  int h = blockIdx.y;            // 0..NH-1
  int i = threadIdx.x;           // 0..63 (pair index)
  float k0v, k1v, v0v, v1v;
  if (p < PREFIX) {
    const float* ks = kc + ((size_t)h * PREFIX + p) * HD;
    const float* vs = vc + ((size_t)h * PREFIX + p) * HD;
    k0v = ks[i]; k1v = ks[i + HD / 2];
    v0v = vs[i]; v1v = vs[i + HD / 2];
  } else {
    int s = p - PREFIX, kvh = h >> 1;  // group = NH/NKVH = 2
    const float* ks = knew + (size_t)s * KVDIM + kvh * HD;
    const float* vs = vnew + (size_t)s * KVDIM + kvh * HD;
    k0v = ks[i]; k1v = ks[i + HD / 2];
    v0v = vs[i]; v1v = vs[i + HD / 2];
  }
  float invf = expf(-(2.f * (float)i / (float)HD) * LOG1E4);
  float sn, cs;
  sincosf((float)p * invf, &sn, &cs);
  float* kd = keys + ((size_t)h * KV_LEN + p) * HD;
  float* vd = vals + ((size_t)h * KV_LEN + p) * HD;
  kd[i] = k0v * cs - k1v * sn;
  kd[i + HD / 2] = k1v * cs + k0v * sn;
  vd[i] = v0v;
  vd[i + HD / 2] = v1v;
}

// ======================================================================
// Flash attention: 1 wave per (16-query tile, head).
//   QK^T and P@V via V_WMMA_F32_16X16X4_F32; online softmax;
//   mask by loop bounds (full prefix + intra-chunk causal).
// ======================================================================
__global__ void __launch_bounds__(32) attn_kernel(
    const float* __restrict__ q, const float* __restrict__ keys,
    const float* __restrict__ vals, bf16* __restrict__ out_hi,
    bf16* __restrict__ out_lo, const int* __restrict__ chunk_ptr) {
  int qt = blockIdx.x;               // 0..S/16-1
  int h  = blockIdx.y;               // head
  int q0 = qt * 16;
  int l = threadIdx.x, lm = l & 15, lhalf = l >> 4;
  int CS = chunk_ptr[0];             // chunk_size (512)
  __shared__ float ldsP[16 * 16];    // P transposed: ldsP[key][query]

  v2f qf[HD / 4];
  const float* qbase = q + (size_t)(q0 + lm) * QDIM + h * HD;
  #pragma unroll
  for (int t = 0; t < HD / 4; t++)
    qf[t] = *(const v2f*)(qbase + 4 * t + 2 * lhalf);

  v8f o[HD / 16] = {};
  float mS[8], lS[8];
  #pragma unroll
  for (int r = 0; r < 8; r++) { mS[r] = -3.0e38f; lS[r] = 0.f; }
  const float scale = 0.08838834764831845f;  // 1/sqrt(HD)

  int c = q0 / CS;
  int localStart = PREFIX + c * CS;
  int diagBlock  = PREFIX + q0;

  for (int phase = 0; phase < 2; phase++) {
    int jb_begin = phase ? localStart : 0;
    int jb_end   = phase ? (diagBlock + 16) : PREFIX;
    for (int jb = jb_begin; jb < jb_end; jb += 16) {
      v8f sacc = {};
      const float* kb = keys + ((size_t)h * KV_LEN + jb + lm) * HD + 2 * lhalf;
      #pragma unroll
      for (int t = 0; t < HD / 4; t++) {
        v2f bf_ = *(const v2f*)(kb + 4 * t);
        sacc = WMMA_F32(qf[t], bf_, sacc);
      }
      bool diag = (jb == diagBlock);
      #pragma unroll
      for (int r = 0; r < 8; r++) {
        float v = sacc[r] * scale;
        if (diag && (lm > r + 8 * lhalf)) v = -3.0e38f;
        sacc[r] = v;
      }
      float pr[8];
      #pragma unroll
      for (int r = 0; r < 8; r++) {
        float v = sacc[r];
        #pragma unroll
        for (int off = 1; off < 16; off <<= 1) v = fmaxf(v, __shfl_xor(v, off, 32));
        float mnew = fmaxf(mS[r], v);
        float corr = __expf(mS[r] - mnew);
        float p = __expf(sacc[r] - mnew);
        float rs = p;
        #pragma unroll
        for (int off = 1; off < 16; off <<= 1) rs += __shfl_xor(rs, off, 32);
        lS[r] = lS[r] * corr + rs;
        mS[r] = mnew;
        pr[r] = p;
        #pragma unroll
        for (int t = 0; t < HD / 16; t++) o[t][r] *= corr;
      }
      #pragma unroll
      for (int r = 0; r < 8; r++)
        ldsP[lm * 16 + r + 8 * lhalf] = pr[r];
      __syncthreads();
      #pragma unroll
      for (int tt = 0; tt < 4; tt++) {
        v2f pa;
        pa.x = ldsP[(4 * tt + 2 * lhalf + 0) * 16 + lm];
        pa.y = ldsP[(4 * tt + 2 * lhalf + 1) * 16 + lm];
        const float* vb = vals + ((size_t)h * KV_LEN + jb + 4 * tt + 2 * lhalf) * HD + lm;
        #pragma unroll
        for (int td = 0; td < HD / 16; td++) {
          v2f bv;
          bv.x = vb[16 * td];
          bv.y = vb[16 * td + HD];
          o[td] = WMMA_F32(pa, bv, o[td]);
        }
      }
      __syncthreads();
    }
  }
  #pragma unroll
  for (int td = 0; td < HD / 16; td++)
    #pragma unroll
    for (int r = 0; r < 8; r++) {
      float v = o[td][r] / lS[r];
      int m = r + 8 * lhalf;
      int dd = 16 * td + lm;
      size_t idx = (size_t)(q0 + m) * QDIM + h * HD + dd;
      bf16 hi = (bf16)v;
      out_hi[idx] = hi;
      out_lo[idx] = (bf16)(v - (float)hi);
    }
}

// ======================================================================
// Host-side launch
// ======================================================================
extern "C" void kernel_launch(void* const* d_in, const int* in_sizes, int n_in,
                              void* d_out, int out_size, void* d_ws, size_t ws_size,
                              hipStream_t stream) {
  (void)in_sizes; (void)n_in; (void)out_size; (void)ws_size;
  const float* hidden  = (const float*)d_in[0];
  const float* k_cache = (const float*)d_in[1];
  const float* v_cache = (const float*)d_in[2];
  const float* q_w     = (const float*)d_in[3];
  const float* k_w     = (const float*)d_in[4];
  const float* v_w     = (const float*)d_in[5];
  const float* o_w     = (const float*)d_in[6];
  const float* gate_w  = (const float*)d_in[7];
  const float* up_w    = (const float*)d_in[8];
  const float* down_w  = (const float*)d_in[9];
  const float* ln1_w   = (const float*)d_in[10];
  const float* ln2_w   = (const float*)d_in[11];
  const int*   chunk   = (const int*)d_in[12];
  float* out = (float*)d_out;

  // Region A is time-multiplexed: QKV splits -> keys/vals -> O split ->
  // gate+up splits -> down split (all lifetime-disjoint).
  char* ws = (char*)d_ws;
  size_t off = 0;
  auto alloc = [&](size_t bytes) { size_t o = off; off += (bytes + 255) & ~(size_t)255; return o; };

  const size_t QW_N  = (size_t)QDIM * D_MODEL;    // 4M
  const size_t KW_N  = (size_t)KVDIM * D_MODEL;   // 2M
  const size_t FFW_N = (size_t)FF_DIM * D_MODEL;  // 11.5M

  size_t o_A   = alloc(4 * FFW_N * 2);                    // 92.3 MB region
  size_t o_hn  = alloc((size_t)S_LEN * D_MODEL * 2 * 2);  // hi+lo (also h2n)
  size_t o_q   = alloc((size_t)S_LEN * QDIM * 4);
  size_t o_kn  = alloc((size_t)S_LEN * KVDIM * 4);
  size_t o_vn  = alloc((size_t)S_LEN * KVDIM * 4);
  size_t o_at  = alloc((size_t)S_LEN * QDIM * 2 * 2);     // hi+lo
  size_t o_h2  = alloc((size_t)S_LEN * D_MODEL * 4);
  size_t o_ff  = alloc((size_t)S_LEN * FF_DIM * 2 * 2);   // hi+lo

  char* A = ws + o_A;
  bf16*  hn_hi = (bf16*)(ws + o_hn);
  bf16*  hn_lo = hn_hi + (size_t)S_LEN * D_MODEL;
  float* qbuf  = (float*)(ws + o_q);
  float* knew  = (float*)(ws + o_kn);
  float* vnew  = (float*)(ws + o_vn);
  bf16*  at_hi = (bf16*)(ws + o_at);
  bf16*  at_lo = at_hi + (size_t)S_LEN * QDIM;
  float* h2    = (float*)(ws + o_h2);
  bf16*  ff_hi = (bf16*)(ws + o_ff);
  bf16*  ff_lo = ff_hi + (size_t)S_LEN * FF_DIM;

  // 1. RMSNorm 1 + split
  rmsnorm_split_kernel<<<S_LEN, 256, 0, stream>>>(hidden, ln1_w, hn_hi, hn_lo, D_MODEL);

  // 2. split QKV weights (region A phase 1)
  bf16* wq_hi = (bf16*)A;
  bf16* wq_lo = wq_hi + QW_N;
  bf16* wk_hi = wq_lo + QW_N;
  bf16* wk_lo = wk_hi + KW_N;
  bf16* wv_hi = wk_lo + KW_N;
  bf16* wv_lo = wv_hi + KW_N;
  split_f32_kernel<<<(QW_N / 8 + 255) / 256, 256, 0, stream>>>(q_w, wq_hi, wq_lo, QW_N);
  split_f32_kernel<<<(KW_N / 8 + 255) / 256, 256, 0, stream>>>(k_w, wk_hi, wk_lo, KW_N);
  split_f32_kernel<<<(KW_N / 8 + 255) / 256, 256, 0, stream>>>(v_w, wv_hi, wv_lo, KW_N);

  // 3. Q/K/V projections
  gemm_bf16s_kernel<<<dim3(QDIM / 128, S_LEN / 64), 256, 0, stream>>>(
      hn_hi, hn_lo, wq_hi, wq_lo, nullptr, qbuf, S_LEN, QDIM, D_MODEL);
  gemm_bf16s_kernel<<<dim3(KVDIM / 128, S_LEN / 64), 256, 0, stream>>>(
      hn_hi, hn_lo, wk_hi, wk_lo, nullptr, knew, S_LEN, KVDIM, D_MODEL);
  gemm_bf16s_kernel<<<dim3(KVDIM / 128, S_LEN / 64), 256, 0, stream>>>(
      hn_hi, hn_lo, wv_hi, wv_lo, nullptr, vnew, S_LEN, KVDIM, D_MODEL);

  // 4. RoPE + KV assembly (region A phase 2)
  float* keys = (float*)A;
  float* vals = keys + (size_t)NH * KV_LEN * HD;
  rope_q_kernel<<<S_LEN, 256, 0, stream>>>(qbuf);
  build_kv_kernel<<<dim3(KV_LEN, NH), 64, 0, stream>>>(k_cache, v_cache, knew, vnew, keys, vals);

  // 5. Flash attention
  attn_kernel<<<dim3(S_LEN / 16, NH), 32, 0, stream>>>(qbuf, keys, vals, at_hi, at_lo, chunk);

  // 6. O projection + residual (region A phase 3)
  bf16* wo_hi = (bf16*)A;
  bf16* wo_lo = wo_hi + QW_N;
  split_f32_kernel<<<(QW_N / 8 + 255) / 256, 256, 0, stream>>>(o_w, wo_hi, wo_lo, QW_N);
  gemm_bf16s_kernel<<<dim3(D_MODEL / 128, S_LEN / 64), 256, 0, stream>>>(
      at_hi, at_lo, wo_hi, wo_lo, hidden, h2, S_LEN, D_MODEL, QDIM);

  // 7. RMSNorm 2 + split (aliases hn)
  rmsnorm_split_kernel<<<S_LEN, 256, 0, stream>>>(h2, ln2_w, hn_hi, hn_lo, D_MODEL);

  // 8. gate/up + SiLU*mul (region A phase 4)
  bf16* wg_hi = (bf16*)A;
  bf16* wg_lo = wg_hi + FFW_N;
  bf16* wu_hi = wg_lo + FFW_N;
  bf16* wu_lo = wu_hi + FFW_N;
  split_f32_kernel<<<(FFW_N / 8 + 255) / 256, 256, 0, stream>>>(gate_w, wg_hi, wg_lo, FFW_N);
  split_f32_kernel<<<(FFW_N / 8 + 255) / 256, 256, 0, stream>>>(up_w, wu_hi, wu_lo, FFW_N);
  gemm_gateup_kernel<<<dim3(FF_DIM / 128, S_LEN / 64), 256, 0, stream>>>(
      hn_hi, hn_lo, wg_hi, wg_lo, wu_hi, wu_lo, ff_hi, ff_lo, S_LEN, FF_DIM, D_MODEL);

  // 9. down projection + residual -> out (region A phase 5)
  bf16* wd_hi = (bf16*)A;
  bf16* wd_lo = wd_hi + FFW_N;
  split_f32_kernel<<<(FFW_N / 8 + 255) / 256, 256, 0, stream>>>(down_w, wd_hi, wd_lo, FFW_N);
  gemm_bf16s_kernel<<<dim3(D_MODEL / 128, S_LEN / 64), 256, 0, stream>>>(
      ff_hi, ff_lo, wd_hi, wd_lo, h2, out, S_LEN, D_MODEL, FF_DIM);
}
